// MultiLayerGAT_61847529062741
// MI455X (gfx1250) — compile-verified, matrix operand order checked by reading, not run on hardware
//
#include <hip/hip_runtime.h>
#include <hip/hip_bf16.h>

typedef __attribute__((ext_vector_type(16))) _Float16 v16h;
typedef __attribute__((ext_vector_type(8)))  float    v8f;

#define NEG_SLOPE 0.2f
#define T 256
#define MAX_K 256
#define LDS_STRIDE (MAX_K + 16)   // halves; 272*2B = 544B row stride (32B multiple)

// ---- ordered-uint encoding for float atomicMax ----
__device__ __forceinline__ unsigned enc_ord(float f) {
  unsigned u = __float_as_uint(f);
  return (u >> 31) ? ~u : (u | 0x80000000u);
}
__device__ __forceinline__ float dec_ord(unsigned u) {
  return (u >> 31) ? __uint_as_float(u & 0x7fffffffu) : __uint_as_float(~u);
}
__device__ __forceinline__ float lrelu(float v) { return v > 0.f ? v : NEG_SLOPE * v; }

// ---------------- utility kernels ----------------
__global__ void gat_zero(unsigned* p, long n) {
  long i = (long)blockIdx.x * blockDim.x + threadIdx.x;
  if (i < n) p[i] = 0u;
}

__global__ void gat_cvt_f16(const float* in, _Float16* out, long n) {
  long i = (long)blockIdx.x * blockDim.x + threadIdx.x;
  if (i < n) out[i] = (_Float16)in[i];
}

// ---------------- WMMA GEMM:  C[M,Nout] = A[M,K] * B[Nout,K]^T ----------------
// A, B in f16 (B stored weight-style: row o holds K contiguous values).
// Block = 8 waves; all waves share one 16-row A tile staged in LDS (loaded
// once instead of 8x). Each wave computes one 16x16 f32 tile via
// v_wmma_f32_16x16x32_f16. The B fragment is double-buffered with the last
// iteration peeled, so the in-loop global_load_b128 is unconditional and
// overlaps the XDL op with no zero-fill merge.
// Requires M%16==0, K%32==0, K<=MAX_K, Nout%128==0 (grid.y = Nout/128).
__global__ void gat_wmma_gemm(const _Float16* __restrict__ A,
                              const _Float16* __restrict__ B,
                              float* __restrict__ C,
                              int M, int K, int Nout) {
  __shared__ _Float16 As[16 * LDS_STRIDE];
  const int tid = threadIdx.x;
  const int r = blockIdx.x * 16;                 // tile row base

  // stage A tile (16 x K halves) into LDS in 16B chunks; uniform trip count
  const int chunksPerRow = K >> 3;               // 8 halves per 16B chunk
  const int totalChunks = 16 * chunksPerRow;
  for (int ci = tid; ci < totalChunks; ci += T) {
    int row = ci / chunksPerRow;
    int w   = ci - row * chunksPerRow;
    *(uint4*)(&As[row * LDS_STRIDE + w * 8]) =
        *(const uint4*)(A + (long)(r + row) * K + w * 8);
  }
  __syncthreads();

  const int wave = tid >> 5;
  const int lane = tid & 31;
  const int c = (blockIdx.y * 8 + wave) * 16;    // tile col base
  const int lr = lane & 15;                      // row (A) / col (B) within tile
  const int hi = lane >> 4;                      // which 16-half chunk of K
  const _Float16* bp  = B + (long)(c + lr) * K + hi * 16;
  const _Float16* apl = &As[lr * LDS_STRIDE + hi * 16];

  v8f acc = {};
  v16h b = *(const v16h*)(bp);                   // prime B double-buffer
  int k = 0;
  for (; k + 32 < K; k += 32) {                  // steady state: unconditional
    __builtin_prefetch(bp + k + 96, 0, 3);       // global_prefetch_b8 (near)
    v16h bn = *(const v16h*)(bp + k + 32);       // next B fragment in flight
    v16h a  = *(const v16h*)(apl + k);           // ds_load_b128 x2
    acc = __builtin_amdgcn_wmma_f32_16x16x32_f16(
        /*neg_a=*/false, a, /*neg_b=*/false, b,
        /*c_mod=*/(short)0, acc, /*reuse_a=*/false, /*reuse_b=*/false);
    b = bn;
  }
  {                                              // peeled final iteration
    v16h a = *(const v16h*)(apl + k);
    acc = __builtin_amdgcn_wmma_f32_16x16x32_f16(
        false, a, false, b, (short)0, acc, false, false);
  }
  // D layout: lanes 0-15 -> M=v, lanes 16-31 -> M=v+8; N = lane%16
#pragma unroll
  for (int v = 0; v < 8; ++v)
    C[(long)(r + v + 8 * hi) * Nout + c + lr] = acc[v];
}

// ---------------- attention logits: wave per (node, head) ----------------
__global__ void gat_att_logits(const float* __restrict__ hp,
                               const float* __restrict__ a_s,
                               const float* __restrict__ a_d,
                               float* als, float* ald, int Nn, int H, int Cdim) {
  int wid = (int)(((long)blockIdx.x * blockDim.x + threadIdx.x) >> 5);
  int lane = threadIdx.x & 31;
  if (wid >= Nn * H) return;
  int h = wid % H, n = wid / H;
  const float* hr = hp + (long)n * H * Cdim + h * Cdim;
  float ss = 0.f, sd = 0.f;
  for (int j = lane; j < Cdim; j += 32) {
    float v = hr[j];
    ss += v * a_s[h * Cdim + j];
    sd += v * a_d[h * Cdim + j];
  }
  for (int off = 16; off; off >>= 1) {
    ss += __shfl_xor(ss, off, 32);
    sd += __shfl_xor(sd, off, 32);
  }
  if (lane == 0) { als[n * H + h] = ss; ald[n * H + h] = sd; }
}

// ---------------- edge pass 1: segment max (thread per (edge, head)) ----------------
__global__ void gat_edge_max(const int* __restrict__ src, const int* __restrict__ dst,
                             int E, long total, int H,
                             const float* __restrict__ als, const float* __restrict__ ald,
                             unsigned* menc) {
  long t = (long)blockIdx.x * blockDim.x + threadIdx.x;
  if (t >= total) return;
  int h = (int)(t % H); long e = t / H;
  int s = (e < E) ? src[e] : (int)(e - E);
  int d = (e < E) ? dst[e] : (int)(e - E);
  float lg = lrelu(als[(long)s * H + h] + ald[(long)d * H + h]);
  atomicMax(&menc[(long)d * H + h], enc_ord(lg));
}

// ---------------- edge pass 2: exp-sum ----------------
__global__ void gat_edge_sum(const int* __restrict__ src, const int* __restrict__ dst,
                             int E, long total, int H,
                             const float* __restrict__ als, const float* __restrict__ ald,
                             const unsigned* __restrict__ menc, float* den) {
  long t = (long)blockIdx.x * blockDim.x + threadIdx.x;
  if (t >= total) return;
  int h = (int)(t % H); long e = t / H;
  int s = (e < E) ? src[e] : (int)(e - E);
  int d = (e < E) ? dst[e] : (int)(e - E);
  float lg = lrelu(als[(long)s * H + h] + ald[(long)d * H + h]);
  float ex = __expf(lg - dec_ord(menc[(long)d * H + h]));
  atomicAdd(&den[(long)d * H + h], ex);
}

// ---------------- edge pass 3: weighted scatter (wave per (edge, head)) ----------------
__global__ void gat_edge_scatter(const int* __restrict__ src, const int* __restrict__ dst,
                                 int E, long total, int H,
                                 const float* __restrict__ als, const float* __restrict__ ald,
                                 const unsigned* __restrict__ menc,
                                 const float* __restrict__ den,
                                 const float* __restrict__ hp, float* out, int Cdim) {
  long wid = ((long)blockIdx.x * blockDim.x + threadIdx.x) >> 5;
  int lane = threadIdx.x & 31;
  if (wid >= total) return;
  int h = (int)(wid % H); long e = wid / H;
  int s = (e < E) ? src[e] : (int)(e - E);
  int d = (e < E) ? dst[e] : (int)(e - E);
  float lg = lrelu(als[(long)s * H + h] + ald[(long)d * H + h]);
  float alpha = __expf(lg - dec_ord(menc[(long)d * H + h])) / den[(long)d * H + h];
  const float* hs = hp + (long)s * H * Cdim + h * Cdim;
  float* od = out + (long)d * H * Cdim + h * Cdim;
  for (int j = lane; j < Cdim; j += 32)
    atomicAdd(&od[j], hs[j] * alpha);
}

// ---------------- bias + relu + f16 repack (layer1 output -> layer2 input) ----------------
__global__ void gat_finalize_relu_f16(const float* __restrict__ acc,
                                      const float* __restrict__ bias,
                                      _Float16* out16, long n, int Cdim) {
  long i = (long)blockIdx.x * blockDim.x + threadIdx.x;
  if (i >= n) return;
  float v = acc[i] + bias[i % Cdim];
  out16[i] = (_Float16)(v > 0.f ? v : 0.f);
}

// ---------------- bias add in place (layer2 output) ----------------
__global__ void gat_finalize_bias(float* acc, const float* __restrict__ bias,
                                  long n, int Cdim) {
  long i = (long)blockIdx.x * blockDim.x + threadIdx.x;
  if (i >= n) return;
  acc[i] += bias[i % Cdim];
}

// ---------------- global mean pool: wave per node ----------------
__global__ void gat_pool(const float* __restrict__ h, const int* __restrict__ batch,
                         float* gsum, float* gcnt, int Nn, int Cdim) {
  int wid = (int)(((long)blockIdx.x * blockDim.x + threadIdx.x) >> 5);
  int lane = threadIdx.x & 31;
  if (wid >= Nn) return;
  int g = batch[wid];
  const float* hr = h + (long)wid * Cdim;
  for (int j = lane; j < Cdim; j += 32)
    atomicAdd(&gsum[(long)g * Cdim + j], hr[j]);
  if (lane == 0) atomicAdd(&gcnt[g], 1.0f);
}

// ---------------- prediction head ----------------
__global__ void gat_head(const float* __restrict__ gsum, const float* __restrict__ gcnt,
                         const float* __restrict__ Wp, const float* __restrict__ bp,
                         float* out, int Gn, int Cdim, int O) {
  int t = blockIdx.x * blockDim.x + threadIdx.x;
  if (t >= Gn * O) return;
  int g = t / O, o = t % O;
  float inv = 1.0f / fmaxf(gcnt[g], 1.0f);
  float s = 0.f;
  for (int k = 0; k < Cdim; ++k)
    s += gsum[(long)g * Cdim + k] * inv * Wp[o * Cdim + k];
  out[t] = s + bp[o];
}

extern "C" void kernel_launch(void* const* d_in, const int* in_sizes, int n_in,
                              void* d_out, int out_size, void* d_ws, size_t ws_size,
                              hipStream_t stream) {
  const float* x    = (const float*)d_in[0];
  const int*   ei   = (const int*)d_in[1];
  const int*   batch= (const int*)d_in[2];
  const float* W1   = (const float*)d_in[3];
  const float* a_s1 = (const float*)d_in[4];
  const float* a_d1 = (const float*)d_in[5];
  const float* b1   = (const float*)d_in[6];
  const float* W2   = (const float*)d_in[7];
  const float* a_s2 = (const float*)d_in[8];
  const float* a_d2 = (const float*)d_in[9];
  const float* b2   = (const float*)d_in[10];
  const float* Wp   = (const float*)d_in[11];
  const float* bp   = (const float*)d_in[12];
  float* out = (float*)d_out;

  const int FIN = 256, HID = 128, H1 = 2, HC1 = 256, OUT = 10;
  const int Nn = in_sizes[0] / FIN;     // 20000
  const int E  = in_sizes[1] / 2;       // 640000
  const int E2 = E + Nn;                // with self loops
  const int Gn = out_size / OUT;        // 64
  const int* src = ei;
  const int* dst = ei + E;

  // ---- workspace carve-out (256B aligned slices) ----
  size_t off = 0;
  auto wsa = [&](size_t bytes) -> void* {
    void* p = (void*)((char*)d_ws + off);
    off += (bytes + 255) & ~(size_t)255;
    return p;
  };
  // zero-initialized region first
  unsigned* m1   = (unsigned*)wsa((size_t)Nn * H1 * 4);
  float*    den1 = (float*)   wsa((size_t)Nn * H1 * 4);
  float*    out1 = (float*)   wsa((size_t)Nn * HC1 * 4);
  unsigned* m2   = (unsigned*)wsa((size_t)Nn * 4);
  float*    den2 = (float*)   wsa((size_t)Nn * 4);
  float*    out2 = (float*)   wsa((size_t)Nn * HID * 4);
  float*    gsum = (float*)   wsa((size_t)Gn * HID * 4);
  float*    gcnt = (float*)   wsa((size_t)Gn * 4);
  const size_t zero_bytes = off;
  // non-zeroed scratch
  _Float16* x16   = (_Float16*)wsa((size_t)Nn * FIN * 2);
  _Float16* w116  = (_Float16*)wsa((size_t)HC1 * FIN * 2);
  _Float16* w216  = (_Float16*)wsa((size_t)HID * HC1 * 2);
  float*    h1    = (float*)   wsa((size_t)Nn * HC1 * 4);
  _Float16* h1r16 = (_Float16*)wsa((size_t)Nn * HC1 * 2);
  float*    h2    = (float*)   wsa((size_t)Nn * HID * 4);
  float*    als1  = (float*)   wsa((size_t)Nn * H1 * 4);
  float*    ald1  = (float*)   wsa((size_t)Nn * H1 * 4);
  float*    als2  = (float*)   wsa((size_t)Nn * 4);
  float*    ald2  = (float*)   wsa((size_t)Nn * 4);
  (void)ws_size; (void)n_in;

  // ---- 0) zero accumulators ----
  long zw = (long)(zero_bytes / 4);
  gat_zero<<<(unsigned)((zw + T - 1) / T), T, 0, stream>>>((unsigned*)d_ws, zw);

  // ---- 1) f32 -> f16 packs ----
  long nx = (long)Nn * FIN;
  gat_cvt_f16<<<(unsigned)((nx + T - 1) / T), T, 0, stream>>>(x, x16, nx);
  gat_cvt_f16<<<(unsigned)(((long)HC1 * FIN + T - 1) / T), T, 0, stream>>>(W1, w116, (long)HC1 * FIN);
  gat_cvt_f16<<<(unsigned)(((long)HID * HC1 + T - 1) / T), T, 0, stream>>>(W2, w216, (long)HID * HC1);

  // ---- 2) layer 1 projection: h1 = x @ W1^T  (WMMA, LDS-staged A) ----
  gat_wmma_gemm<<<dim3(Nn / 16, HC1 / 128), T, 0, stream>>>(x16, w116, h1, Nn, FIN, HC1);

  // ---- 3) layer 1 attention + softmax-scatter ----
  {
    long waves = (long)Nn * H1;
    gat_att_logits<<<(unsigned)((waves * 32 + T - 1) / T), T, 0, stream>>>(h1, a_s1, a_d1, als1, ald1, Nn, H1, HID);
    long tot = (long)E2 * H1;
    gat_edge_max<<<(unsigned)((tot + T - 1) / T), T, 0, stream>>>(src, dst, E, tot, H1, als1, ald1, m1);
    gat_edge_sum<<<(unsigned)((tot + T - 1) / T), T, 0, stream>>>(src, dst, E, tot, H1, als1, ald1, m1, den1);
    gat_edge_scatter<<<(unsigned)((tot * 32 + T - 1) / T), T, 0, stream>>>(src, dst, E, tot, H1, als1, ald1, m1, den1, h1, out1, HID);
  }
  // bias + relu + f16 repack for layer 2
  {
    long n = (long)Nn * HC1;
    gat_finalize_relu_f16<<<(unsigned)((n + T - 1) / T), T, 0, stream>>>(out1, b1, h1r16, n, HC1);
  }

  // ---- 4) layer 2 projection: h2 = relu(out1) @ W2^T  (WMMA, LDS-staged A) ----
  gat_wmma_gemm<<<dim3(Nn / 16, HID / 128), T, 0, stream>>>(h1r16, w216, h2, Nn, HC1, HID);

  // ---- 5) layer 2 attention + softmax-scatter (H=1) ----
  {
    long waves = (long)Nn;
    gat_att_logits<<<(unsigned)((waves * 32 + T - 1) / T), T, 0, stream>>>(h2, a_s2, a_d2, als2, ald2, Nn, 1, HID);
    long tot = (long)E2;
    gat_edge_max<<<(unsigned)((tot + T - 1) / T), T, 0, stream>>>(src, dst, E, tot, 1, als2, ald2, m2);
    gat_edge_sum<<<(unsigned)((tot + T - 1) / T), T, 0, stream>>>(src, dst, E, tot, 1, als2, ald2, m2, den2);
    gat_edge_scatter<<<(unsigned)((tot * 32 + T - 1) / T), T, 0, stream>>>(src, dst, E, tot, 1, als2, ald2, m2, den2, h2, out2, HID);
  }
  {
    long n = (long)Nn * HID;
    gat_finalize_bias<<<(unsigned)((n + T - 1) / T), T, 0, stream>>>(out2, b2, n, HID);
  }

  // ---- 6) mean pool + head ----
  gat_pool<<<(unsigned)(((long)Nn * 32 + T - 1) / T), T, 0, stream>>>(out2, batch, gsum, gcnt, Nn, HID);
  gat_head<<<(unsigned)((Gn * OUT + T - 1) / T), T, 0, stream>>>(gsum, gcnt, Wp, bp, out, Gn, HID, OUT);
}
